// K_25847113187694
// MI455X (gfx1250) — matrix operand order repaired and checked by reading, not compiled, not run on hardware
//
#include <hip/hip_runtime.h>

// ---------------------------------------------------------------------------
// Class-conditioned batched GEMM:  out[b] = x[b] (64x1152) @ W[y[b]] (1152x128) + bias[y[b]]
// MI455X (gfx1250): wave32, WMMA 16x16x32 bf16, f32 accumulate, bf16x3 split
// for ~f32 accuracy while staying on the memory-bound roofline (~40us @ 23.3TB/s).
// ---------------------------------------------------------------------------

#define HID    1152
#define NROW   64
#define ODIM   128
#define KC     32
#define NCHUNK (HID / KC)      // 36

// LDS layout (per buffer): bf16, padded pitch 80B (=40 bf16) to avoid bank conflicts
#define XPITCH 80
#define XHI    0               // x hi  : 64 rows * 80B  = 5120
#define XLO    5120            // x lo  : 5120
#define WHI    10240           // w hi  : 128 rows * 80B = 10240  (stored transposed [N][K])
#define WLO    20480           // w lo  : 10240
#define BUFSZ  30720           // per buffer; double buffered -> 60 KB LDS

typedef __attribute__((ext_vector_type(16))) __bf16       bf16x16;
typedef __attribute__((ext_vector_type(2)))  __bf16       bf16x2;
typedef __attribute__((ext_vector_type(8)))  float        f32x8;
typedef __attribute__((ext_vector_type(4)))  unsigned int u32x4;
typedef __attribute__((ext_vector_type(4)))  float        f32x4v;

union Frag { bf16x16 v; u32x4 q[2]; };   // trivial types only -> union is well-formed

// split two f32 into packed (hi,hi) and (lo,lo) bf16 dwords.
// Written with native casts so the backend can use v_cvt_pk_bf16_f32 if present;
// LLVM's expansion is the standard RNE bit sequence otherwise.
__device__ __forceinline__ void splitpk(float a, float b, unsigned &hi, unsigned &lo) {
    bf16x2 h;
    h[0] = (__bf16)a;
    h[1] = (__bf16)b;
    hi = __builtin_bit_cast(unsigned, h);
    float ra = a - (float)h[0];
    float rb = b - (float)h[1];
    bf16x2 l;
    l[0] = (__bf16)ra;
    l[1] = (__bf16)rb;
    lo = __builtin_bit_cast(unsigned, l);
}

__global__ __launch_bounds__(256, 1) void classcond_gemm_kernel(
    const float* __restrict__ x,     // [B, 64, 1152]
    const int*   __restrict__ y,     // [B]
    const float* __restrict__ w,     // [1000, 1152, 128]
    const float* __restrict__ bias,  // [1000, 128]
    float*       __restrict__ out)   // [B, 64, 128]
{
    __shared__ alignas(16) unsigned char lds[2 * BUFSZ];

    const int b    = blockIdx.x;
    const int tid  = threadIdx.x;
    const int lane = tid & 31;
    const int wv   = tid >> 5;            // wave id 0..7 -> N tile
    const int cls  = y[b];

    const float* xb = x + (size_t)b   * NROW * HID;
    const float* wb = w + (size_t)cls * HID  * ODIM;

    // ---- staging roles (all 256 threads) ----
    const int xr_row = tid >> 2;          // 0..63 : x row
    const int xr_q   = tid & 3;           // 0..3  : 8 consecutive K values
    const int wn     = tid & 127;         // 0..127: W output column (coalesced per k-row)
    const int wh     = tid >> 7;          // 0..1  : K half (16 values)

    float xr[8];
    float wr[16];

    auto gload = [&](int kk) {
        // x chunk: row xr_row, K = kk*32 + xr_q*8 .. +7   (two 16B loads)
        const float* px = xb + (size_t)xr_row * HID + kk * KC + xr_q * 8;
        const f32x4v v0 = *(const f32x4v*)(px);
        const f32x4v v1 = *(const f32x4v*)(px + 4);
        xr[0]=v0[0]; xr[1]=v0[1]; xr[2]=v0[2]; xr[3]=v0[3];
        xr[4]=v1[0]; xr[5]=v1[1]; xr[6]=v1[2]; xr[7]=v1[3];
        // W chunk column wn, K = kk*32 + wh*16 .. +15 (each k-row read is 128B coalesced)
        const float* pw = wb + (size_t)(kk * KC + wh * 16) * ODIM + wn;
        #pragma unroll
        for (int i = 0; i < 16; ++i) wr[i] = pw[(size_t)i * ODIM];
    };

    auto cstore = [&](int buf) {
        unsigned char* base = lds + buf * BUFSZ;
        // x -> bf16 hi/lo, row-major [M][K] (A-fragment layout)
        unsigned xh[4], xl[4];
        #pragma unroll
        for (int i = 0; i < 4; ++i) splitpk(xr[2*i], xr[2*i+1], xh[i], xl[i]);
        u32x4 th = {xh[0], xh[1], xh[2], xh[3]};
        u32x4 tl = {xl[0], xl[1], xl[2], xl[3]};
        *(u32x4*)(base + XHI + xr_row * XPITCH + xr_q * 16) = th;
        *(u32x4*)(base + XLO + xr_row * XPITCH + xr_q * 16) = tl;
        // W -> bf16 hi/lo, transposed [N][K] (B-fragment layout: per-lane K column)
        unsigned wh8[8], wl8[8];
        #pragma unroll
        for (int i = 0; i < 8; ++i) splitpk(wr[2*i], wr[2*i+1], wh8[i], wl8[i]);
        u32x4 wh0 = {wh8[0], wh8[1], wh8[2], wh8[3]};
        u32x4 wh1 = {wh8[4], wh8[5], wh8[6], wh8[7]};
        u32x4 wl0 = {wl8[0], wl8[1], wl8[2], wl8[3]};
        u32x4 wl1 = {wl8[4], wl8[5], wl8[6], wl8[7]};
        unsigned char* ph = base + WHI + wn * XPITCH + wh * 32;
        unsigned char* pl = base + WLO + wn * XPITCH + wh * 32;
        ((u32x4*)ph)[0] = wh0;  ((u32x4*)ph)[1] = wh1;
        ((u32x4*)pl)[0] = wl0;  ((u32x4*)pl)[1] = wl1;
    };

    // ---- accumulators: 4 M-tiles of 16x16 per wave ----
    f32x8 acc[4];
    #pragma unroll
    for (int mt = 0; mt < 4; ++mt) acc[mt] = (f32x8)0.0f;

    const int koff = lane >> 4;           // 0: K lo half lanes, 1: K hi half lanes
    const int nr   = lane & 15;

    gload(0);
    for (int kk = 0; kk < NCHUNK; ++kk) {
        const int buf = kk & 1;
        cstore(buf);                      // convert prefetched chunk into LDS[buf]
        if (kk + 1 < NCHUNK) gload(kk + 1);   // issue next global loads (overlap WMMA)
        __syncthreads();

        unsigned char* base = lds + buf * BUFSZ;

        // B fragments (W), reused across the 4 M tiles.
        // Layout: lanes 0-15 hold K=0..15 of col N=lane; lanes 16-31 K=16..31 of N=lane-16.
        unsigned char* pbh = base + WHI + (wv * 16 + nr) * XPITCH + koff * 32;
        unsigned char* pbl = base + WLO + (wv * 16 + nr) * XPITCH + koff * 32;
        Frag bh, bl;
        bh.q[0] = ((const u32x4*)pbh)[0];  bh.q[1] = ((const u32x4*)pbh)[1];
        bl.q[0] = ((const u32x4*)pbl)[0];  bl.q[1] = ((const u32x4*)pbl)[1];

        #pragma unroll
        for (int mt = 0; mt < 4; ++mt) {
            // A fragment: lanes<16 row M=lane K{0..15}; lanes>=16 row M=lane-16 K{16..31}
            unsigned char* pah = base + XHI + (mt * 16 + nr) * XPITCH + koff * 32;
            unsigned char* pal = base + XLO + (mt * 16 + nr) * XPITCH + koff * 32;
            Frag ah, al;
            ah.q[0] = *(const u32x4*)(pah);  ah.q[1] = *(const u32x4*)(pah + 16);
            al.q[0] = *(const u32x4*)(pal);  al.q[1] = *(const u32x4*)(pal + 16);

            // bf16x3: hi*hi + hi*lo + lo*hi  (f32 accumulate, ~f32 accuracy)
            acc[mt] = __builtin_amdgcn_wmma_f32_16x16x32_bf16(
                false, ah.v, false, bh.v, (short)0, acc[mt], false, false);
            acc[mt] = __builtin_amdgcn_wmma_f32_16x16x32_bf16(
                false, ah.v, false, bl.v, (short)0, acc[mt], false, false);
            acc[mt] = __builtin_amdgcn_wmma_f32_16x16x32_bf16(
                false, al.v, false, bh.v, (short)0, acc[mt], false, false);
        }
    }

    // ---- epilogue: bias add + store ----
    // C/D layout: VGPR j holds row M = j + 8*(lane>=16), col N = lane%16 (within tile)
    const int   ncol = wv * 16 + nr;
    const float bv   = bias[(size_t)cls * ODIM + ncol];
    float* ob = out + (size_t)b * NROW * ODIM;
    const int mbase = koff * 8;
    #pragma unroll
    for (int mt = 0; mt < 4; ++mt) {
        #pragma unroll
        for (int j = 0; j < 8; ++j) {
            ob[(size_t)(mt * 16 + mbase + j) * ODIM + ncol] = acc[mt][j] + bv;
        }
    }
}

extern "C" void kernel_launch(void* const* d_in, const int* in_sizes, int n_in,
                              void* d_out, int out_size, void* d_ws, size_t ws_size,
                              hipStream_t stream) {
    const float* x    = (const float*)d_in[0];
    const int*   y    = (const int*)  d_in[1];   // jnp int64 w/o x64 -> int32
    const float* w    = (const float*)d_in[2];
    const float* bias = (const float*)d_in[3];
    float*       out  = (float*)d_out;

    const int B = out_size / (NROW * ODIM);      // 1024
    classcond_gemm_kernel<<<dim3(B), dim3(256), 0, stream>>>(x, y, w, bias, out);
}
